// GatedMLP_56839597195714
// MI455X (gfx1250) — compile-verified, matrix operand order checked by reading, not run on hardware
//
#include <hip/hip_runtime.h>

#define DEV __device__ __forceinline__

typedef __attribute__((ext_vector_type(16))) __bf16 v16bf;
typedef __attribute__((ext_vector_type(8)))  float  v8f;
typedef __attribute__((ext_vector_type(4)))  int    v4i;
typedef __attribute__((address_space(1))) v4i gv4i;   // global int4
typedef __attribute__((address_space(3))) v4i lv4i;   // LDS int4

// ---- problem constants (from the reference) ----
static constexpr int NS    = 4096;   // N samples
static constexpr int INDIM = 1024;   // IN_DIM
static constexpr int DD    = 2048;   // D
static constexpr int TWOD  = 4096;   // 2*D
static constexpr int NG    = 8;      // gate count
static constexpr int OUTD  = 1024;   // OUT_DIM
static constexpr float GSCALE = 0.9f;

// ---- workspace layout (bytes) ----
static constexpr size_t OFF_XB  = 0;                                         // x  bf16 [NS,INDIM]
static constexpr size_t OFF_W1B = OFF_XB  + (size_t)NS * INDIM * 2;          // W1 bf16 [(1+G)*2D, INDIM]
static constexpr size_t OFF_W2B = OFF_W1B + (size_t)(1 + NG) * TWOD * INDIM * 2;
static constexpr size_t OFF_W3B = OFF_W2B + (size_t)DD * TWOD * 2;           // W2 bf16 [DD,TWOD]
static constexpr size_t OFF_S   = OFF_W3B + (size_t)OUTD * DD * 2;           // W3 bf16 [OUTD,DD]
static constexpr size_t OFF_H2  = OFF_S   + (size_t)NS * TWOD * 2;           // s  bf16 [NS,TWOD]
static constexpr size_t OFF_LST = OFF_H2  + (size_t)NS * DD * 2;             // h2 bf16 [NS,DD]
static constexpr size_t OFF_CNT = OFF_LST + (size_t)NG * NS * 4;             // lists int [G,NS]

// ---- small helpers ----
DEV unsigned short f2bf(float f) {
  union { float f; unsigned u; } x; x.f = f;
  unsigned r = x.u + 0x7FFFu + ((x.u >> 16) & 1u);   // round-to-nearest-even
  return (unsigned short)(r >> 16);
}
DEV float bf2f(unsigned short u) {
  union { unsigned u; float f; } x; x.u = ((unsigned)u) << 16;
  return x.f;
}

struct alignas(16) U4 { unsigned x, y, z, w; };
struct alignas(16) F4 { float x, y, z, w; };
struct alignas(8)  U2 { unsigned x, y; };

union FragB16 {            // 16 bf16 = one WMMA A/B fragment per lane
  v16bf v;
  U4    q[2];
};

// ---- async global -> LDS copy (CDNA5 DMA path, ASYNCcnt-tracked) ----
#if defined(__gfx1250__) && __has_builtin(__builtin_amdgcn_global_load_async_to_lds_b128)
#define USE_ASYNC_LDS 1
#else
#define USE_ASYNC_LDS 0
#endif

DEV void copy_b128_to_lds(const unsigned short* gsrc, unsigned short* ldst) {
#if USE_ASYNC_LDS
  __builtin_amdgcn_global_load_async_to_lds_b128(
      (gv4i*)(unsigned short*)gsrc,   // const dropped, AS1 int4*
      (lv4i*)ldst,                    // AS3 int4*
      0, 0);
#else
  *(U4*)ldst = *(const U4*)gsrc;
#endif
}

DEV void wait_async_copies() {
#if USE_ASYNC_LDS
#if __has_builtin(__builtin_amdgcn_s_wait_asynccnt)
  __builtin_amdgcn_s_wait_asynccnt(0);
#else
  asm volatile("s_wait_asynccnt 0x0" ::: "memory");
#endif
#endif
}

// ---------------- fp32 -> bf16 conversion (vectorized x4) ----------------
__global__ void cvt_f32_bf16(const float* __restrict__ src,
                             unsigned short* __restrict__ dst, int n4) {
  int i = blockIdx.x * blockDim.x + threadIdx.x;
  int stride = gridDim.x * blockDim.x;
  for (; i < n4; i += stride) {
    F4 f = ((const F4*)src)[i];
    U2 p;
    p.x = (unsigned)f2bf(f.x) | ((unsigned)f2bf(f.y) << 16);
    p.y = (unsigned)f2bf(f.z) | ((unsigned)f2bf(f.w) << 16);
    ((U2*)dst)[i] = p;
  }
}

// ------------- deterministic bucket-by-gate (8 waves, 1 block) -------------
__global__ void build_groups(const int* __restrict__ gids,
                             int* __restrict__ lists,
                             int* __restrict__ counts, int n, int ncap) {
  int w = threadIdx.x >> 5;        // wave = gate id (wave32)
  int lane = threadIdx.x & 31;
  if (w >= NG) return;
  int* mylist = lists + (size_t)w * ncap;
  int cnt = 0;
  for (int base = 0; base < n; base += 32) {
    int id = gids[base + lane];
    bool match = (id == w);
    unsigned m = (unsigned)__ballot(match);
    int pos = __popc(m & ((1u << lane) - 1u));
    if (match) mylist[cnt + pos] = base + lane;
    cnt += __popc(m);
  }
  if (lane == 0) counts[w] = cnt;
}

// --------------------------- WMMA bf16 GEMM -------------------------------
// C[M,Nc] = epilogue(A[M,K] @ B[Nc,K]^T + bias)
// Block tile 64(M) x 128(N), 128 threads = 4 waves (2 M-waves x 2 N-waves),
// each wave computes 32x64 via 2 A-frags x 4 B-frags = 8 WMMA per K-step
// (each B fragment feeds 2 WMMAs -> 1.5 LDS b128 loads per WMMA).
// Double-buffered LDS; global->LDS via async DMA (ASYNCcnt) when available.
// EPI 0: out bf16 = relu(.)                (shared layer-1 block)
// EPI 1: gathered rows; out bf16 = bf2f(out) + gscale*relu(.)   (gate blocks)
// EPI 2: out bf16 = relu(.)                (layer 2)
// EPI 3: out f32  = (.)  no bias/relu      (layer 3)
template <int EPI>
__global__ __launch_bounds__(128)
void gemm_bf16_wmma(const unsigned short* __restrict__ A,
                    const unsigned short* __restrict__ Bw,
                    const float* __restrict__ bias,
                    void* __restrict__ outp,
                    const int* __restrict__ lists,
                    const int* __restrict__ counts,
                    int M, int Nc, int K, int ldout, float gscale, int ncap) {
  __shared__ unsigned short As[2][64][72];    // +8 shorts pad (16B) per row
  __shared__ unsigned short Bs[2][128][72];
  __shared__ int rowmap[64];

  const int tid  = threadIdx.x;
  const int lane = tid & 31;               // wave32
  const int wv   = tid >> 5;               // 0..3
  const int wmv  = wv & 1;                 // 2 waves down (M), 32 rows each
  const int wnv  = wv >> 1;                // 2 waves across (N), 64 cols each
  const int half = lane >> 4;              // lane group (ISA A/B frag layout)

  const int m0 = blockIdx.x * 64;
  const int n0 = blockIdx.y * 128;

  int cnt = M;
  const unsigned short* B = Bw;
  const float* bs = bias;
  if constexpr (EPI == 1) {
    const int g = blockIdx.z;
    cnt = counts[g];
    if (m0 >= cnt) return;                 // block-uniform: EXEC stays full
    B  = Bw   + (size_t)g * TWOD * K;
    bs = bias + (size_t)g * TWOD;
    const int* list = lists + (size_t)g * ncap;
    if (tid < 64) {
      int mi = m0 + tid;
      if (mi >= cnt) mi = cnt - 1;         // clamp; masked in epilogue
      rowmap[tid] = list[mi];
    }
  } else {
    if (tid < 64) rowmap[tid] = m0 + tid;
  }
  __syncthreads();

  // --- loop-invariant copy assignments (128 thr): A = 4 chunks, B = 8 ---
  const int cr = tid >> 3;                 // chunk row 0..15
  const int cc = (tid & 7) * 8;            // chunk col (shorts)
  const unsigned short* srcA[4];
  const unsigned short* srcB[8];
#pragma unroll
  for (int j = 0; j < 4; ++j)
    srcA[j] = A + (size_t)rowmap[cr + 16 * j] * K + cc;
#pragma unroll
  for (int j = 0; j < 8; ++j)
    srcB[j] = B + (size_t)(n0 + cr + 16 * j) * K + cc;

  v8f acc[2][4] = {};

  const int KT = K >> 6;                   // 64-wide K tiles

#define COPY_TILE(kt, buf)                                                   \
  do {                                                                       \
    const int ko = (kt) * 64;                                                \
    _Pragma("unroll")                                                        \
    for (int j = 0; j < 4; ++j)                                              \
      copy_b128_to_lds(srcA[j] + ko, &As[buf][cr + 16 * j][cc]);             \
    _Pragma("unroll")                                                        \
    for (int j = 0; j < 8; ++j)                                              \
      copy_b128_to_lds(srcB[j] + ko, &Bs[buf][cr + 16 * j][cc]);             \
  } while (0)

  COPY_TILE(0, 0);

  const int ar0 = wmv * 32 + (lane & 15);  // first 16-row A sub-tile
  const int ar1 = ar0 + 16;                // second 16-row A sub-tile
  const int br  = wnv * 64 + (lane & 15);

  for (int kt = 0; kt < KT; ++kt) {
    const int buf = kt & 1;
    wait_async_copies();   // own async copies of tile kt have landed in LDS
    __syncthreads();       // all waves' copies landed; all done with buf^1
    if (kt + 1 < KT) COPY_TILE(kt + 1, buf ^ 1);   // overlaps with compute

#pragma unroll
    for (int ks = 0; ks < 2; ++ks) {       // two 16x16x32 K-steps per stage
      const int col = ks * 32 + half * 8;  // ISA bf16 frag: lanes 16-31 -> K+8
      FragB16 a0, a1;
      a0.q[0] = *(const U4*)&As[buf][ar0][col];
      a0.q[1] = *(const U4*)&As[buf][ar0][col + 16];
      a1.q[0] = *(const U4*)&As[buf][ar1][col];
      a1.q[1] = *(const U4*)&As[buf][ar1][col + 16];
#pragma unroll
      for (int t = 0; t < 4; ++t) {
        FragB16 b;
        b.q[0] = *(const U4*)&Bs[buf][br + t * 16][col];
        b.q[1] = *(const U4*)&Bs[buf][br + t * 16][col + 16];
        acc[0][t] = __builtin_amdgcn_wmma_f32_16x16x32_bf16(
            false, a0.v, false, b.v, (short)0, acc[0][t], false, false);
        acc[1][t] = __builtin_amdgcn_wmma_f32_16x16x32_bf16(
            false, a1.v, false, b.v, (short)0, acc[1][t], false, false);
      }
    }
  }
#undef COPY_TILE

  // ---- epilogue ----  C layout: m = v + 8*half, n = lane&15 (per VGPR v)
  const int nbase = n0 + wnv * 64 + (lane & 15);
#pragma unroll
  for (int sm = 0; sm < 2; ++sm) {
    const int mloc = wmv * 32 + sm * 16 + 8 * half;
#pragma unroll
    for (int t = 0; t < 4; ++t) {
      const int n = nbase + t * 16;
      float bb = 0.f;
      if constexpr (EPI != 3) bb = bs[n];
#pragma unroll
      for (int v = 0; v < 8; ++v) {
        const int ml = mloc + v;
        float val = acc[sm][t][v] + bb;
        if constexpr (EPI == 0 || EPI == 2) {
          val = val > 0.f ? val : 0.f;
          ((unsigned short*)outp)[(size_t)(m0 + ml) * ldout + n] = f2bf(val);
        } else if constexpr (EPI == 1) {
          if (m0 + ml < cnt) {
            unsigned short* o = (unsigned short*)outp;
            const int row = rowmap[ml];
            float prev = bf2f(o[(size_t)row * ldout + n]);
            val = val > 0.f ? val : 0.f;
            o[(size_t)row * ldout + n] = f2bf(prev + gscale * val);
          }
        } else {
          ((float*)outp)[(size_t)(m0 + ml) * ldout + n] = val;
        }
      }
    }
  }
}

// ------------------------------- launcher ---------------------------------
extern "C" void kernel_launch(void* const* d_in, const int* in_sizes, int n_in,
                              void* d_out, int out_size, void* d_ws, size_t ws_size,
                              hipStream_t stream) {
  (void)in_sizes; (void)n_in; (void)out_size; (void)ws_size;
  const float* x   = (const float*)d_in[0];
  const int*   gid = (const int*)d_in[1];
  const float* W1  = (const float*)d_in[2];
  const float* b1  = (const float*)d_in[3];
  const float* W2  = (const float*)d_in[4];
  const float* b2  = (const float*)d_in[5];
  const float* W3  = (const float*)d_in[6];
  float* out = (float*)d_out;

  char* ws = (char*)d_ws;
  unsigned short* xb  = (unsigned short*)(ws + OFF_XB);
  unsigned short* w1b = (unsigned short*)(ws + OFF_W1B);
  unsigned short* w2b = (unsigned short*)(ws + OFF_W2B);
  unsigned short* w3b = (unsigned short*)(ws + OFF_W3B);
  unsigned short* sb  = (unsigned short*)(ws + OFF_S);
  unsigned short* h2b = (unsigned short*)(ws + OFF_H2);
  int* lists  = (int*)(ws + OFF_LST);
  int* counts = (int*)(ws + OFF_CNT);

  const dim3 cvtblk(256);
  const dim3 gemmblk(128);

  // 1) fp32 -> bf16 for all GEMM operands
  cvt_f32_bf16<<<4096, cvtblk, 0, stream>>>(x,  xb,  NS * INDIM / 4);
  cvt_f32_bf16<<<4096, cvtblk, 0, stream>>>(W1, w1b, (1 + NG) * TWOD * INDIM / 4);
  cvt_f32_bf16<<<4096, cvtblk, 0, stream>>>(W2, w2b, DD * TWOD / 4);
  cvt_f32_bf16<<<4096, cvtblk, 0, stream>>>(W3, w3b, OUTD * DD / 4);

  // 2) bucket samples by gate id (deterministic ballot/prefix compaction)
  build_groups<<<1, cvtblk, 0, stream>>>(gid, lists, counts, NS, NS);

  // 3) layer-1 shared block:  s = bf16(relu(x @ W1_shared^T + b1_shared))
  gemm_bf16_wmma<0><<<dim3(NS / 64, TWOD / 128, 1), gemmblk, 0, stream>>>(
      xb, w1b + (size_t)NG * TWOD * INDIM, b1 + (size_t)NG * TWOD, sb,
      nullptr, nullptr, NS, TWOD, INDIM, TWOD, 0.f, NS);

  // 4) layer-1 gate blocks (gathered rows per gate, in-place combine into s)
  gemm_bf16_wmma<1><<<dim3(NS / 64, TWOD / 128, NG), gemmblk, 0, stream>>>(
      xb, w1b, b1, sb, lists, counts, NS, TWOD, INDIM, TWOD, GSCALE, NS);

  // 5) layer-2:  h2 = bf16(relu(s @ W2^T + b2))
  gemm_bf16_wmma<2><<<dim3(NS / 64, DD / 128, 1), gemmblk, 0, stream>>>(
      sb, w2b, b2, h2b, nullptr, nullptr, NS, DD, TWOD, DD, 0.f, NS);

  // 6) layer-3:  out(f32) = h2 @ W3^T
  gemm_bf16_wmma<3><<<dim3(NS / 64, OUTD / 128, 1), gemmblk, 0, stream>>>(
      h2b, w3b, nullptr, out, nullptr, nullptr, NS, OUTD, DD, OUTD, 0.f, NS);
}
